// CausalSelfAttention_74337293959583
// MI455X (gfx1250) — matrix operand order repaired and checked by reading, not compiled, not run on hardware
//
#include <hip/hip_runtime.h>

typedef __attribute__((ext_vector_type(16))) _Float16 v16h;
typedef __attribute__((ext_vector_type(8)))  float    v8f;
typedef __attribute__((ext_vector_type(4)))  unsigned int v4u;
typedef __attribute__((ext_vector_type(8)))  int      v8i;
typedef __attribute__((ext_vector_type(4)))  int      v4i;

#define D_MODEL 1024
#define N_HEADS 16
#define SEQ     2048
#define BATCH   2
#define HD      64
#define MTOT    (BATCH * SEQ)   // 4096 rows

union AFrag { v16h h; unsigned u[8]; };

__device__ __forceinline__ v8f wmma_f16(const v16h& a, const v16h& b, const v8f& c) {
    return __builtin_amdgcn_wmma_f32_16x16x32_f16(false, a, false, b, (short)0, c, false, false);
}

__device__ __forceinline__ unsigned lds_off(const void* p) {
    // generic LDS pointer: low 32 bits are the wave-relative LDS byte offset
    return (unsigned)(unsigned long long)(size_t)p;
}

// ---------------------------------------------------------------------------
// Tensor Data Mover: 2-D tile load Global->LDS (D# per ISA 08_async_tensor §8)
//   data_size = 2 bytes, pad enabled:
//   pad_interval code c -> 2^(c+1) dwords between pads; pad_amount code a -> a+1 dwords
// 6-arg builtin form (clang-23): (g0, g1, g2, g3, extra, cpol)
// ---------------------------------------------------------------------------
__device__ __forceinline__ void tdm_load_2d(unsigned lds_addr, unsigned long long gaddr,
                                            unsigned tdim0, unsigned tdim1,
                                            unsigned long long stride0,
                                            unsigned tile0, unsigned tile1,
                                            unsigned pad_interval, unsigned pad_amount)
{
    v4u g0;
    g0[0] = 1u;                                            // count=1, user descriptor
    g0[1] = lds_addr;                                      // [63:32] lds_addr
    g0[2] = (unsigned)(gaddr & 0xffffffffull);             // [95:64] global_addr lo
    g0[3] = (unsigned)((gaddr >> 32) & 0x1ffffffull)       // [120:96] global_addr hi
          | (2u << 30);                                    // [127:126] type=2 (image)
    v8i g1;
    g1[0] = (int)((1u << 16)                               // data_size = 2 bytes
                | (1u << 20)                               // pad_enable
                | (pad_interval << 22)
                | (pad_amount << 25));
    g1[1] = (int)((tdim0 & 0xffffu) << 16);                // tensor_dim0[15:0] (bar addr=0)
    g1[2] = (int)(((tdim0 >> 16) & 0xffffu) | ((tdim1 & 0xffffu) << 16));
    g1[3] = (int)(((tdim1 >> 16) & 0xffffu) | ((tile0 & 0xffffu) << 16));
    g1[4] = (int)(tile1 & 0xffffu);                        // tile_dim1, tile_dim2=0
    g1[5] = (int)(stride0 & 0xffffffffull);                // tensor_dim0_stride lo
    g1[6] = (int)((stride0 >> 32) & 0xffffu);              // stride hi, dim1_stride=0
    g1[7] = 0;
    v4i z4 = {0, 0, 0, 0};
    v8i z8 = {0, 0, 0, 0, 0, 0, 0, 0};
    __builtin_amdgcn_tensor_load_to_lds(g0, g1, z4, z4, z8, 0);
}

// ---------------------------------------------------------------------------
// Pre-pass: f32 -> f16 convert (x), and transpose+convert (weights -> [N][K])
// ---------------------------------------------------------------------------
__global__ __launch_bounds__(256)
void csa_cvt_f16(const float* __restrict__ in, _Float16* __restrict__ outh) {
    size_t idx = ((size_t)blockIdx.x * 256 + threadIdx.x) * 4;
    float4 v = *(const float4*)(in + idx);
    union { _Float16 h[4]; unsigned long long u; } p;
    p.h[0] = (_Float16)v.x; p.h[1] = (_Float16)v.y;
    p.h[2] = (_Float16)v.z; p.h[3] = (_Float16)v.w;
    *(unsigned long long*)(outh + idx) = p.u;
}

__global__ __launch_bounds__(256)
void csa_transpose_w(const float* __restrict__ W, _Float16* __restrict__ Wt) {
    __shared__ _Float16 ts[64][65];
    const int tid = threadIdx.x;
    const int r0 = blockIdx.y * 64, c0 = blockIdx.x * 64;
#pragma unroll
    for (int i = 0; i < 16; ++i) {
        int lin = tid + i * 256;
        int r = lin >> 6, c = lin & 63;
        ts[r][c] = (_Float16)W[(size_t)(r0 + r) * D_MODEL + c0 + c];
    }
    __syncthreads();
#pragma unroll
    for (int i = 0; i < 16; ++i) {
        int lin = tid + i * 256;
        int r = lin >> 6, c = lin & 63;
        Wt[(size_t)(c0 + r) * D_MODEL + r0 + c] = ts[c][r];
    }
}

// ---------------------------------------------------------------------------
// GEMM: C[M,N] = A[f16, MxK] @ Wt[f16, NxK]^T + bias   (M=4096, N=K=1024)
// MODE 0: f16 out, [B,H,S,hd]   (Q, K)
// MODE 1: f16 out, [B,H,hd,S]   (V transposed)
// MODE 2: f32 out, row-major    (final projection -> d_out)
// Tiles staged by TDM, double-buffered; 8 waves (2x4), 128x128x32.
// ---------------------------------------------------------------------------
template<int MODE>
__global__ __launch_bounds__(256)
void csa_gemm_kernel(const _Float16* __restrict__ A, const _Float16* __restrict__ Wt,
                     const float* __restrict__ bias, void* __restrict__ out)
{
    constexpr int K = D_MODEL, N = D_MODEL;
    __shared__ _Float16 sA[2][128][36];   // TDM pad: 64B row + 8B pad = 72B stride
    __shared__ _Float16 sB[2][128][36];

    const int tid   = threadIdx.x;
    const int lane  = tid & 31;
    const int wid   = tid >> 5;
    const int waveM = wid & 1;
    const int waveN = wid >> 1;
    const int nl    = lane & 15;
    const int kg    = lane >> 4;
    const int m0    = blockIdx.y * 128;
    const int n0    = blockIdx.x * 128;

    const unsigned long long aBase = (unsigned long long)(size_t)A;
    const unsigned long long wBase = (unsigned long long)(size_t)Wt;

    if (wid == 0) {   // wave 0 drives the Tensor Data Mover
        tdm_load_2d(lds_off(&sA[0][0][0]), aBase + 2ull * ((size_t)m0 * K), K, MTOT, K, 32, 128, 3, 1);
        tdm_load_2d(lds_off(&sB[0][0][0]), wBase + 2ull * ((size_t)n0 * K), K, N,    K, 32, 128, 3, 1);
    }

    v8f acc[4][2];
#pragma unroll
    for (int mt = 0; mt < 4; ++mt)
#pragma unroll
        for (int nt = 0; nt < 2; ++nt)
            acc[mt][nt] = (v8f){0.f,0.f,0.f,0.f,0.f,0.f,0.f,0.f};

    constexpr int kiter = K / 32;
    for (int it = 0; it < kiter; ++it) {
        const int buf = it & 1;
        if (wid == 0) {
            if (it + 1 < kiter) {
                const int k0n = (it + 1) * 32;
                tdm_load_2d(lds_off(&sA[buf ^ 1][0][0]), aBase + 2ull * ((size_t)m0 * K + k0n), K, MTOT, K, 32, 128, 3, 1);
                tdm_load_2d(lds_off(&sB[buf ^ 1][0][0]), wBase + 2ull * ((size_t)n0 * K + k0n), K, N,    K, 32, 128, 3, 1);
                __builtin_amdgcn_s_wait_tensorcnt(2);   // current buffer complete
            } else {
                __builtin_amdgcn_s_wait_tensorcnt(0);
            }
        }
        __syncthreads();

        AFrag a[4], b[2];
#pragma unroll
        for (int mt = 0; mt < 4; ++mt) {
            int row = waveM * 64 + mt * 16 + nl;
#pragma unroll
            for (int v = 0; v < 8; ++v) {
                int k = kg * 8 + (v & 3) * 2 + (v >> 2) * 16;
                a[mt].u[v] = *(const unsigned*)&sA[buf][row][k];
            }
        }
#pragma unroll
        for (int nt = 0; nt < 2; ++nt) {
            int col = waveN * 32 + nt * 16 + nl;
#pragma unroll
            for (int v = 0; v < 8; ++v) {
                int k = kg * 8 + (v & 3) * 2 + (v >> 2) * 16;
                b[nt].u[v] = *(const unsigned*)&sB[buf][col][k];
            }
        }
#pragma unroll
        for (int mt = 0; mt < 4; ++mt)
#pragma unroll
            for (int nt = 0; nt < 2; ++nt)
                acc[mt][nt] = wmma_f16(a[mt].h, b[nt].h, acc[mt][nt]);
        __syncthreads();
    }

#pragma unroll
    for (int mt = 0; mt < 4; ++mt) {
#pragma unroll
        for (int nt = 0; nt < 2; ++nt) {
            int n = n0 + waveN * 32 + nt * 16 + nl;
            float bv = bias[n];
#pragma unroll
            for (int r = 0; r < 8; ++r) {
                int m = m0 + waveM * 64 + mt * 16 + r + kg * 8;
                float val = acc[mt][nt][r] + bv;
                if constexpr (MODE == 2) {
                    ((float*)out)[(size_t)m * N + n] = val;
                } else {
                    int b_ = m >> 11, s_ = m & (SEQ - 1);
                    int h_ = n >> 6,  d_ = n & (HD - 1);
                    if constexpr (MODE == 0)
                        ((_Float16*)out)[(((size_t)(b_ * N_HEADS + h_) * SEQ) + s_) * HD + d_] = (_Float16)val;
                    else
                        ((_Float16*)out)[(((size_t)(b_ * N_HEADS + h_) * HD) + d_) * SEQ + s_] = (_Float16)val;
                }
            }
        }
    }
}

// ---------------------------------------------------------------------------
// Flash attention (causal). Grid = (S/64, B*H), block = 128 = 4 waves.
// K/V tiles staged once per key-block by TDM (shared by all waves), double-
// buffered across the kb loop. Q fragments stay resident in VGPRs.
// ---------------------------------------------------------------------------
__global__ __launch_bounds__(128)
void csa_flash_kernel(const _Float16* __restrict__ Q, const _Float16* __restrict__ Kx,
                      const _Float16* __restrict__ Vx, _Float16* __restrict__ Ot)
{
    __shared__ _Float16 sK[2][64][72];   // [key][d], TDM pad: 128B row + 16B = 144B stride
    __shared__ _Float16 sV[2][64][72];   // [d][key]
    __shared__ _Float16 sP[4][16][68];   // wave-private P re-layout slices

    const int tid  = threadIdx.x;
    const int lane = tid & 31;
    const int w    = tid >> 5;
    const int nl   = lane & 15;
    const int kg   = lane >> 4;
    const int qb   = blockIdx.x;
    const int bh   = blockIdx.y;
    const int b    = bh >> 4;
    const int h    = bh & 15;
    const int m0   = qb * 64;

    const _Float16* qp = Q + (size_t)bh * SEQ * HD;
    const unsigned long long kBase = (unsigned long long)(size_t)(Kx + (size_t)bh * SEQ * HD);
    const unsigned long long vBase = (unsigned long long)(size_t)(Vx + (size_t)bh * HD * SEQ);

    // Q A-fragments (rows m0 + w*16 + [0,16)), resident all kernel
    AFrag qf[2];
    {
        int row = m0 + w * 16 + nl;
#pragma unroll
        for (int kk = 0; kk < 2; ++kk)
#pragma unroll
            for (int v = 0; v < 8; ++v) {
                int d = kk * 32 + kg * 8 + (v & 3) * 2 + (v >> 2) * 16;
                qf[kk].u[v] = *(const unsigned*)(qp + (size_t)row * HD + d);
            }
    }

    if (w == 0) {   // prologue: stage key-block 0
        tdm_load_2d(lds_off(&sK[0][0][0]), kBase, HD, SEQ, HD, 64, 64, 4, 3);
        tdm_load_2d(lds_off(&sV[0][0][0]), vBase, SEQ, HD, SEQ, 64, 64, 4, 3);
    }

    v8f oacc[4];
#pragma unroll
    for (int nt = 0; nt < 4; ++nt) oacc[nt] = (v8f){0.f,0.f,0.f,0.f,0.f,0.f,0.f,0.f};
    float mstat[8], lstat[8];
#pragma unroll
    for (int r = 0; r < 8; ++r) { mstat[r] = -__builtin_inff(); lstat[r] = 0.f; }

    const float scale = 0.125f;   // 1/sqrt(64)

    for (int kb = 0; kb <= qb; ++kb) {
        const int j0  = kb * 64;
        const int buf = kb & 1;

        if (w == 0) {
            if (kb + 1 <= qb) {
                const int j0n = (kb + 1) * 64;
                tdm_load_2d(lds_off(&sK[buf ^ 1][0][0]), kBase + 2ull * (size_t)j0n * HD, HD, SEQ, HD, 64, 64, 4, 3);
                tdm_load_2d(lds_off(&sV[buf ^ 1][0][0]), vBase + 2ull * (size_t)j0n,     SEQ, HD, SEQ, 64, 64, 4, 3);
                __builtin_amdgcn_s_wait_tensorcnt(2);
            } else {
                __builtin_amdgcn_s_wait_tensorcnt(0);
            }
        }
        __syncthreads();

        // ---- S = Q * K^T (16 x 64 per wave), K^T B-frags from LDS --------
        v8f sacc[4];
#pragma unroll
        for (int nt = 0; nt < 4; ++nt) {
            sacc[nt] = (v8f){0.f,0.f,0.f,0.f,0.f,0.f,0.f,0.f};
            int col = nt * 16 + nl;   // local key index
#pragma unroll
            for (int kk = 0; kk < 2; ++kk) {
                AFrag bf;
#pragma unroll
                for (int v = 0; v < 8; ++v) {
                    int d = kk * 32 + kg * 8 + (v & 3) * 2 + (v >> 2) * 16;
                    bf.u[v] = *(const unsigned*)&sK[buf][col][d];
                }
                sacc[nt] = wmma_f16(qf[kk].h, bf.h, sacc[nt]);
            }
        }

        // ---- scale + causal mask (diagonal block only) -------------------
        if (kb == qb) {
#pragma unroll
            for (int nt = 0; nt < 4; ++nt)
#pragma unroll
                for (int r = 0; r < 8; ++r) {
                    int qrow = m0 + w * 16 + r + kg * 8;
                    int jcol = j0 + nt * 16 + nl;
                    float sv = sacc[nt][r] * scale;
                    sacc[nt][r] = (jcol > qrow) ? -__builtin_inff() : sv;
                }
        } else {
#pragma unroll
            for (int nt = 0; nt < 4; ++nt)
#pragma unroll
                for (int r = 0; r < 8; ++r) sacc[nt][r] *= scale;
        }

        // ---- row max (lanes 0-15 rows r, lanes 16-31 rows r+8) -----------
        float rmax[8];
#pragma unroll
        for (int r = 0; r < 8; ++r) {
            float v = sacc[0][r];
#pragma unroll
            for (int nt = 1; nt < 4; ++nt) v = fmaxf(v, sacc[nt][r]);
#pragma unroll
            for (int off = 1; off < 16; off <<= 1) v = fmaxf(v, __shfl_xor(v, off, 32));
            rmax[r] = v;
        }

        // ---- online softmax update / rescale O ---------------------------
#pragma unroll
        for (int r = 0; r < 8; ++r) {
            float nm = fmaxf(mstat[r], rmax[r]);
            float alpha = __expf(mstat[r] - nm);
            mstat[r] = nm;
            lstat[r] *= alpha;
#pragma unroll
            for (int nt = 0; nt < 4; ++nt) oacc[nt][r] *= alpha;
        }

        // ---- P = exp(S - m); LDS round-trip for A-fragment layout --------
        float rs[8];
#pragma unroll
        for (int r = 0; r < 8; ++r) rs[r] = 0.f;
#pragma unroll
        for (int nt = 0; nt < 4; ++nt)
#pragma unroll
            for (int r = 0; r < 8; ++r) {
                float p = __expf(sacc[nt][r] - mstat[r]);
                rs[r] += p;
                sP[w][r + kg * 8][nt * 16 + nl] = (_Float16)p;
            }
#pragma unroll
        for (int r = 0; r < 8; ++r) {
            float v = rs[r];
#pragma unroll
            for (int off = 1; off < 16; off <<= 1) v += __shfl_xor(v, off, 32);
            lstat[r] += v;
        }

        // ---- O += P * V ---------------------------------------------------
        AFrag pf[2];
#pragma unroll
        for (int kk = 0; kk < 2; ++kk)
#pragma unroll
            for (int v = 0; v < 8; ++v) {
                int kq = kk * 32 + kg * 8 + (v & 3) * 2 + (v >> 2) * 16;
                pf[kk].u[v] = *(const unsigned*)&sP[w][nl][kq];
            }
#pragma unroll
        for (int nt = 0; nt < 4; ++nt) {
            int col = nt * 16 + nl;   // d column
#pragma unroll
            for (int kk = 0; kk < 2; ++kk) {
                AFrag vf;
#pragma unroll
                for (int v = 0; v < 8; ++v) {
                    int key = kk * 32 + kg * 8 + (v & 3) * 2 + (v >> 2) * 16;   // local key
                    vf.u[v] = *(const unsigned*)&sV[buf][col][key];
                }
                oacc[nt] = wmma_f16(pf[kk].h, vf.h, oacc[nt]);
            }
        }
        __syncthreads();   // release buf for the TDM overwrite next round
    }

    // ---- normalize, store f16 [B,S,D] ------------------------------------
#pragma unroll
    for (int nt = 0; nt < 4; ++nt)
#pragma unroll
        for (int r = 0; r < 8; ++r) {
            int srow = m0 + w * 16 + r + kg * 8;
            int col  = h * HD + nt * 16 + nl;
            Ot[((size_t)b * SEQ + srow) * D_MODEL + col] = (_Float16)(oacc[nt][r] / lstat[r]);
        }
}

// ---------------------------------------------------------------------------
extern "C" void kernel_launch(void* const* d_in, const int* in_sizes, int n_in,
                              void* d_out, int out_size, void* d_ws, size_t ws_size,
                              hipStream_t stream) {
    (void)in_sizes; (void)n_in; (void)out_size; (void)ws_size;
    const float* x  = (const float*)d_in[0];
    const float* wq = (const float*)d_in[1];
    const float* bq = (const float*)d_in[2];
    const float* wk = (const float*)d_in[3];
    const float* bk = (const float*)d_in[4];
    const float* wv = (const float*)d_in[5];
    const float* bv = (const float*)d_in[6];
    const float* wo = (const float*)d_in[7];
    const float* bo = (const float*)d_in[8];

    const size_t MB = 1024ull * 1024ull;
    char* ws = (char*)d_ws;
    _Float16* xh  = (_Float16*)(ws);                // 8 MB  [4096][1024]
    _Float16* wqT = (_Float16*)(ws +  8 * MB);      // 2 MB  [N][K]
    _Float16* wkT = (_Float16*)(ws + 10 * MB);
    _Float16* wvT = (_Float16*)(ws + 12 * MB);
    _Float16* woT = (_Float16*)(ws + 14 * MB);
    _Float16* qh  = (_Float16*)(ws + 16 * MB);      // 8 MB  [B,H,S,hd]
    _Float16* kh  = (_Float16*)(ws + 24 * MB);      // 8 MB  [B,H,S,hd]
    _Float16* vt  = (_Float16*)(ws + 32 * MB);      // 8 MB  [B,H,hd,S]
    _Float16* att = (_Float16*)(ws + 40 * MB);      // 8 MB  [B,S,D]

    // pre-pass: f16 conversion + weight transposes
    csa_cvt_f16<<<dim3((MTOT * D_MODEL) / 1024), 256, 0, stream>>>(x, xh);
    dim3 gT(D_MODEL / 64, D_MODEL / 64);            // (16,16)
    csa_transpose_w<<<gT, 256, 0, stream>>>(wq, wqT);
    csa_transpose_w<<<gT, 256, 0, stream>>>(wk, wkT);
    csa_transpose_w<<<gT, 256, 0, stream>>>(wv, wvT);
    csa_transpose_w<<<gT, 256, 0, stream>>>(wo, woT);

    dim3 gGemm(D_MODEL / 128, MTOT / 128);          // (8, 32)
    csa_gemm_kernel<0><<<gGemm, 256, 0, stream>>>(xh, wqT, bq, (void*)qh);
    csa_gemm_kernel<0><<<gGemm, 256, 0, stream>>>(xh, wkT, bk, (void*)kh);
    csa_gemm_kernel<1><<<gGemm, 256, 0, stream>>>(xh, wvT, bv, (void*)vt);

    dim3 gFlash(SEQ / 64, BATCH * N_HEADS);         // (32, 32)
    csa_flash_kernel<<<gFlash, 128, 0, stream>>>(qh, kh, vt, att);

    csa_gemm_kernel<2><<<gGemm, 256, 0, stream>>>(att, woT, bo, d_out);
}
